// MultiHeadAttention_61272003445327
// MI455X (gfx1250) — matrix operand order repaired and checked by reading, not compiled, not run on hardware
//
#include <hip/hip_runtime.h>

typedef __bf16 bf16;
typedef __attribute__((ext_vector_type(16))) __bf16 v16bf;
typedef __attribute__((ext_vector_type(8)))  __bf16 v8bf;
typedef __attribute__((ext_vector_type(4)))  __bf16 v4bf;
typedef __attribute__((ext_vector_type(8)))  float   v8f;
typedef __attribute__((ext_vector_type(4)))  float   v4f;
typedef __attribute__((ext_vector_type(4)))  unsigned int u32x4;
typedef __attribute__((ext_vector_type(8)))  int      i32x8;
typedef __attribute__((ext_vector_type(4)))  int      i32x4;

#define DEVINL __device__ __forceinline__

// Problem constants
constexpr int Bsz  = 4;
constexpr int Cctx = 2048;
constexpr int Emb  = 1024;
constexpr int NH   = 16;
constexpr int Dh   = 64;
constexpr int HD   = NH * Dh;    // 1024
constexpr int Rows = Bsz * Cctx; // 8192

// ---------------------------------------------------------------------------
// WMMA fragment helpers (layouts per CDNA5 ISA 7.12.2, wave32)
// A (16x32 bf16): lane m=lane&15, h=lane>>4; elem e in 0..7  -> K = 8h+e
//                                          e in 8..15 -> K = 16+8h+(e-8)
// B (32x16 bf16): lane n=lane&15, h=lane>>4; elem e -> K as for A (8h+e, 16+8h+(e-8)), column n
// C/D (16x16 f32): lane n=lane&15, h=lane>>4; VGPR r -> M = r+8h, N = n
// ---------------------------------------------------------------------------

DEVINL v16bf load_a_frag(const bf16* __restrict__ base, int ld, int lane) {
  const int h = lane >> 4, m = lane & 15;
  const bf16* p = base + (size_t)m * ld + h * 8;
  v8bf lo = *(const v8bf*)(p);        // K = 8h .. 8h+7
  v8bf hi = *(const v8bf*)(p + 16);   // K = 16+8h .. 16+8h+7
  return __builtin_shufflevector(lo, hi, 0, 1, 2, 3, 4, 5, 6, 7,
                                          8, 9, 10, 11, 12, 13, 14, 15);
}

DEVINL v16bf load_b_frag(const bf16* __restrict__ base, int ld, int lane) {
  // base points at (n0, k0) of a row-major [N][ld] matrix holding B^T.
  const int h = lane >> 4, n = lane & 15;
  const bf16* p = base + (size_t)n * ld + h * 8;
  v8bf lo = *(const v8bf*)(p);
  v8bf hi = *(const v8bf*)(p + 16);
  return __builtin_shufflevector(lo, hi, 0, 1, 2, 3, 4, 5, 6, 7,
                                          8, 9, 10, 11, 12, 13, 14, 15);
}

DEVINL v8f wmma_bf16(v16bf a, v16bf b, v8f c) {
  return __builtin_amdgcn_wmma_f32_16x16x32_bf16(
      false, a, false, b, (short)0, c, false, false);
}

// ---------------------------------------------------------------------------
// Tensor Data Mover: 2D tile Global -> LDS (D# packing per ISA ch.8.3-8.6)
//   data_size = 2 bytes (code 1); group0.type = 2 ("image"); count = 1
//   tile_dim0 elements per row, tile_dim1 rows, row stride stride0 elements.
//   This toolchain's builtin takes 6 args (g0, g1, g2, g3, g4, cpol).
// ---------------------------------------------------------------------------
DEVINL void tdm_load_2d(unsigned lds_addr, const bf16* gptr,
                        unsigned tile_d0, unsigned tile_d1,
                        unsigned tensor_d0, unsigned tensor_d1,
                        unsigned stride0) {
  const unsigned long long ga = (unsigned long long)gptr;
  u32x4 g0;
  g0[0] = 1u;                                   // count=1, user descriptor
  g0[1] = lds_addr;                             // LDS byte address
  g0[2] = (unsigned)(ga & 0xFFFFFFFFu);         // global_addr[31:0]
  g0[3] = (unsigned)((ga >> 32) & 0x01FFFFFFu)  // global_addr[56:32]
        | (2u << 30);                           // type=2
  i32x8 g1;
  g1[0] = (int)(1u << 16);                      // data_size=1 (2 bytes)
  g1[1] = (int)((tensor_d0 & 0xFFFFu) << 16);   // tensor_dim0[15:0] @bit48
  g1[2] = (int)((tensor_d0 >> 16) | ((tensor_d1 & 0xFFFFu) << 16));
  g1[3] = (int)((tensor_d1 >> 16) | (tile_d0 << 16)); // tile_dim0 @bit112
  g1[4] = (int)(tile_d1 & 0xFFFFu);             // tile_dim1; tile_dim2=0
  g1[5] = (int)stride0;                         // tensor_dim0_stride[31:0]
  g1[6] = 0;                                    // stride hi + dim1_stride lo
  g1[7] = 0;
  i32x4 z4 = {0, 0, 0, 0};                      // groups 2/3 unused (2D)
  i32x8 z8 = {0, 0, 0, 0, 0, 0, 0, 0};
  __builtin_amdgcn_tensor_load_to_lds(g0, g1, z4, z4, z8, 0);
}

// ---------------------------------------------------------------------------
// Kernel 1: x (f32) -> bf16
// ---------------------------------------------------------------------------
__global__ __launch_bounds__(256) void cvt_x_kernel(
    const float* __restrict__ x, bf16* __restrict__ xb, int nvec) {
  int i = blockIdx.x * blockDim.x + threadIdx.x;
  if (i < nvec) {
    v4f f = *(const v4f*)(x + (size_t)i * 4);
    v4bf o;
    o[0] = (bf16)f[0]; o[1] = (bf16)f[1];
    o[2] = (bf16)f[2]; o[3] = (bf16)f[3];
    *(v4bf*)(xb + (size_t)i * 4) = o;
  }
}

// ---------------------------------------------------------------------------
// Kernel 2: transpose + convert a 1024x1024 f32 weight to bf16 Wt[n][k]
// ---------------------------------------------------------------------------
__global__ __launch_bounds__(256) void cvt_w_kernel(
    const float* __restrict__ w0, const float* __restrict__ w1,
    const float* __restrict__ w2, const float* __restrict__ w3,
    bf16* __restrict__ t0, bf16* __restrict__ t1,
    bf16* __restrict__ t2, bf16* __restrict__ t3) {
  __shared__ float tile[16][17];
  const float* W;
  bf16* T;
  switch (blockIdx.z) {
    case 0:  W = w0; T = t0; break;
    case 1:  W = w1; T = t1; break;
    case 2:  W = w2; T = t2; break;
    default: W = w3; T = t3; break;
  }
  const int tx = threadIdx.x & 15, ty = threadIdx.x >> 4;
  const int kbase = blockIdx.x * 16, nbase = blockIdx.y * 16;
  tile[ty][tx] = W[(size_t)(kbase + ty) * Emb + nbase + tx];
  __syncthreads();
  T[(size_t)(nbase + ty) * Emb + kbase + tx] = (bf16)tile[tx][ty];
}

// ---------------------------------------------------------------------------
// Kernel 3: QKV projection. One wave computes a 32x64 output tile
// (2 A-frags x 4 B-frags = 8 WMMAs per k-step; ~10.7 flop/L2-byte).
// ---------------------------------------------------------------------------
__global__ __launch_bounds__(256) void qkv_kernel(
    const bf16* __restrict__ xb,
    const bf16* __restrict__ wqt, const bf16* __restrict__ wkt,
    const bf16* __restrict__ wvt,
    bf16* __restrict__ Q, bf16* __restrict__ K, bf16* __restrict__ Vt) {
  const int lane = threadIdx.x & 31, wave = threadIdx.x >> 5;
  const int h = lane >> 4, n = lane & 15;
  const int mat  = blockIdx.z;
  const int row0 = blockIdx.x * 256 + wave * 32;
  const int col0 = blockIdx.y * 64;
  const bf16* Wt = (mat == 0) ? wqt : ((mat == 1) ? wkt : wvt);

  v8f acc[2][4];
#pragma unroll
  for (int u = 0; u < 2; ++u)
#pragma unroll
    for (int j = 0; j < 4; ++j)
#pragma unroll
      for (int r = 0; r < 8; ++r) acc[u][j][r] = 0.0f;

  const bf16* arow = xb + (size_t)row0 * Emb;
  for (int k0 = 0; k0 < Emb; k0 += 32) {
    __builtin_prefetch(arow + k0 + 128, 0, 0);
    v16bf a0 = load_a_frag(arow + k0, Emb, lane);
    v16bf a1 = load_a_frag(arow + (size_t)16 * Emb + k0, Emb, lane);
#pragma unroll
    for (int j = 0; j < 4; ++j) {
      v16bf bb = load_b_frag(Wt + (size_t)(col0 + j * 16) * Emb + k0, Emb, lane);
      acc[0][j] = wmma_bf16(a0, bb, acc[0][j]);
      acc[1][j] = wmma_bf16(a1, bb, acc[1][j]);
    }
  }

  const float scale = (mat == 0) ? 0.125f : 1.0f; // 1/sqrt(64)
#pragma unroll
  for (int u = 0; u < 2; ++u)
#pragma unroll
    for (int j = 0; j < 4; ++j)
#pragma unroll
      for (int r = 0; r < 8; ++r) {
        const int i  = row0 + u * 16 + r + 8 * h;
        const int jg = col0 + j * 16 + n;
        const int hh = jg >> 6, d = jg & 63;
        const int b  = i >> 11, c = i & 2047;
        const float v = acc[u][j][r] * scale;
        if (mat == 2)
          Vt[(((size_t)b * NH + hh) * Dh + d) * Cctx + c] = (bf16)v;
        else if (mat == 0)
          Q[(((size_t)b * NH + hh) * Cctx + c) * Dh + d] = (bf16)v;
        else
          K[(((size_t)b * NH + hh) * Cctx + c) * Dh + d] = (bf16)v;
      }
}

// ---------------------------------------------------------------------------
// Kernel 4: flash attention with TDM-staged, double-buffered K/V tiles.
// Block = 8 waves = 128 q rows of one (b,h). Wave 0 drives the Tensor Data
// Mover; tiles of 128 keys land in LDS (K: [128][64], V^T: [64][128]).
// Per-wave online softmax; causal bound handled inside the barrier-free
// inner loop so the outer (barriered) loop count is block-uniform.
// ---------------------------------------------------------------------------
__global__ __launch_bounds__(256) void attn_kernel(
    const bf16* __restrict__ Q, const bf16* __restrict__ K,
    const bf16* __restrict__ Vt, bf16* __restrict__ hidden) {
  __shared__ __align__(16) bf16 klds[2][128 * Dh];  // 2 x 16 KB
  __shared__ __align__(16) bf16 vlds[2][Dh * 128];  // 2 x 16 KB
  __shared__ __align__(16) bf16 plds[8][16 * 40];   // per-wave P scratch

  const int lane = threadIdx.x & 31, wave = threadIdx.x >> 5;
  const int h = lane >> 4, n = lane & 15;
  const int b = blockIdx.z, hh = blockIdx.y;
  const int q0blk = blockIdx.x * 128;
  const int q0 = q0blk + wave * 16;

  const bf16* Qb = Q  + ((size_t)(b * NH + hh)) * Cctx * Dh;
  const bf16* Kb = K  + ((size_t)(b * NH + hh)) * Cctx * Dh;
  const bf16* Vb = Vt + ((size_t)(b * NH + hh)) * Dh * Cctx;

  const v16bf aq0 = load_a_frag(Qb + (size_t)q0 * Dh + 0,  Dh, lane);
  const v16bf aq1 = load_a_frag(Qb + (size_t)q0 * Dh + 32, Dh, lane);

  float mrun[8], lrun[8];
  v8f acc[4];
#pragma unroll
  for (int r = 0; r < 8; ++r) { mrun[r] = -3.0e38f; lrun[r] = 0.0f; }
#pragma unroll
  for (int j = 0; j < 4; ++j)
#pragma unroll
    for (int r = 0; r < 8; ++r) acc[j][r] = 0.0f;

  bf16* pl = &plds[wave][0];
  const int ntiles = blockIdx.x + 1; // 128-key tiles up to the diagonal

  // Prologue: stage tile 0 into buffer 0.
  if (wave == 0) {
    tdm_load_2d((unsigned)(size_t)&klds[0][0], Kb, /*tile*/ 128 * Dh, 1,
                /*tensor*/ 128 * Dh, 1, 128 * Dh);           // contiguous
    tdm_load_2d((unsigned)(size_t)&vlds[0][0], Vb, /*tile*/ 128, Dh,
                /*tensor*/ Cctx, Dh, Cctx);                  // strided rows
  }

  for (int it = 0; it < ntiles; ++it) {
    const int kt0 = it * 128;
    if (wave == 0) {
      if (it + 1 < ntiles) {
        const int nk = (it + 1) * 128;
        tdm_load_2d((unsigned)(size_t)&klds[(it + 1) & 1][0],
                    Kb + (size_t)nk * Dh, 128 * Dh, 1, 128 * Dh, 1, 128 * Dh);
        tdm_load_2d((unsigned)(size_t)&vlds[(it + 1) & 1][0],
                    Vb + nk, 128, Dh, Cctx, Dh, Cctx);
        __builtin_amdgcn_s_wait_tensorcnt(2); // tensor ops retire in-order:
                                              // <=2 outstanding => tile `it` done
      } else {
        __builtin_amdgcn_s_wait_tensorcnt(0);
      }
    }
    __syncthreads(); // tile `it` visible to all waves
    const bf16* kbuf = &klds[it & 1][0];
    const bf16* vbuf = &vlds[it & 1][0];

    const int kstop = min(kt0 + 128, q0 + 16); // per-wave causal bound
    for (int k0 = kt0; k0 < kstop; k0 += 32) {
      const int kl = k0 - kt0;
      // ---- S = Q K^T (two 16x16 f32 fragments) ----
      v8f s[2];
#pragma unroll
      for (int t = 0; t < 2; ++t) {
        const bf16* kb = kbuf + (size_t)(kl + t * 16) * Dh;
        v16bf b0 = load_b_frag(kb + 0,  Dh, lane);
        v16bf b1 = load_b_frag(kb + 32, Dh, lane);
        v8f c;
#pragma unroll
        for (int r = 0; r < 8; ++r) c[r] = 0.0f;
        c = wmma_bf16(aq0, b0, c);
        c = wmma_bf16(aq1, b1, c);
        s[t] = c;
      }

      // ---- causal mask + running row max ----
      float mnew[8];
#pragma unroll
      for (int r = 0; r < 8; ++r) {
        const int q = q0 + r + 8 * h;
        if (k0 + n      > q) s[0][r] = -3.0e38f;
        if (k0 + 16 + n > q) s[1][r] = -3.0e38f;
        float mx = fmaxf(s[0][r], s[1][r]);
        for (int off = 1; off < 16; off <<= 1)
          mx = fmaxf(mx, __shfl_xor(mx, off, 32));
        mnew[r] = fmaxf(mrun[r], mx);
      }

      // ---- P = exp(S - mnew), update l, rescale accumulator ----
      float p0[8], p1[8];
#pragma unroll
      for (int r = 0; r < 8; ++r) {
        const float sc = __expf(mrun[r] - mnew[r]);
        p0[r] = __expf(s[0][r] - mnew[r]);
        p1[r] = __expf(s[1][r] - mnew[r]);
        float rs = p0[r] + p1[r];
        for (int off = 1; off < 16; off <<= 1)
          rs += __shfl_xor(rs, off, 32);
        lrun[r] = lrun[r] * sc + rs;
        mrun[r] = mnew[r];
#pragma unroll
        for (int j = 0; j < 4; ++j) acc[j][r] *= sc;
      }

      // ---- re-layout P (C/D layout) -> A-fragment layout via LDS ----
#pragma unroll
      for (int r = 0; r < 8; ++r) {
        pl[(r + 8 * h) * 40 + n]      = (bf16)p0[r];
        pl[(r + 8 * h) * 40 + 16 + n] = (bf16)p1[r];
      }
      __builtin_amdgcn_wave_barrier(); // per-wave LDS ops are in-order
      const bf16* ap = pl + (size_t)n * 40 + h * 8;
      v8bf lo = *(const v8bf*)(ap);
      v8bf hi = *(const v8bf*)(ap + 16);
      v16bf pa = __builtin_shufflevector(lo, hi, 0, 1, 2, 3, 4, 5, 6, 7,
                                                 8, 9, 10, 11, 12, 13, 14, 15);
      __builtin_amdgcn_wave_barrier();

      // ---- O += P @ V ----
#pragma unroll
      for (int j = 0; j < 4; ++j) {
        v16bf bv = load_b_frag(vbuf + (size_t)(j * 16) * 128 + kl, 128, lane);
        acc[j] = wmma_bf16(pa, bv, acc[j]);
      }
    }
    __syncthreads(); // everyone done with buffer (it&1) before it is reused
  }

  // ---- normalize and store hidden [B][C][H*D] bf16 ----
#pragma unroll
  for (int j = 0; j < 4; ++j)
#pragma unroll
    for (int r = 0; r < 8; ++r) {
      const float o = acc[j][r] / lrun[r];
      const size_t c = (size_t)q0 + r + 8 * h;
      hidden[((size_t)b * Cctx + c) * HD + hh * Dh + j * 16 + n] = (bf16)o;
    }
}

// ---------------------------------------------------------------------------
// Kernel 5: output projection, f32 result. One wave -> 32x64 tile.
// ---------------------------------------------------------------------------
__global__ __launch_bounds__(256) void oproj_kernel(
    const bf16* __restrict__ hb, const bf16* __restrict__ wot,
    float* __restrict__ out) {
  const int lane = threadIdx.x & 31, wave = threadIdx.x >> 5;
  const int h = lane >> 4, n = lane & 15;
  const int row0 = blockIdx.x * 256 + wave * 32;
  const int col0 = blockIdx.y * 64;

  v8f acc[2][4];
#pragma unroll
  for (int u = 0; u < 2; ++u)
#pragma unroll
    for (int j = 0; j < 4; ++j)
#pragma unroll
      for (int r = 0; r < 8; ++r) acc[u][j][r] = 0.0f;

  const bf16* arow = hb + (size_t)row0 * HD;
  for (int k0 = 0; k0 < HD; k0 += 32) {
    __builtin_prefetch(arow + k0 + 128, 0, 0);
    v16bf a0 = load_a_frag(arow + k0, HD, lane);
    v16bf a1 = load_a_frag(arow + (size_t)16 * HD + k0, HD, lane);
#pragma unroll
    for (int j = 0; j < 4; ++j) {
      v16bf bb = load_b_frag(wot + (size_t)(col0 + j * 16) * HD + k0, HD, lane);
      acc[0][j] = wmma_bf16(a0, bb, acc[0][j]);
      acc[1][j] = wmma_bf16(a1, bb, acc[1][j]);
    }
  }

#pragma unroll
  for (int u = 0; u < 2; ++u)
#pragma unroll
    for (int j = 0; j < 4; ++j)
#pragma unroll
      for (int r = 0; r < 8; ++r) {
        const size_t i  = (size_t)row0 + u * 16 + r + 8 * h;
        const size_t jg = col0 + j * 16 + n;
        out[i * Emb + jg] = acc[u][j][r];
      }
}

// ---------------------------------------------------------------------------
// Host launcher
// ---------------------------------------------------------------------------
extern "C" void kernel_launch(void* const* d_in, const int* in_sizes, int n_in,
                              void* d_out, int out_size, void* d_ws, size_t ws_size,
                              hipStream_t stream) {
  const float* x  = (const float*)d_in[0];
  const float* wq = (const float*)d_in[1];
  const float* wk = (const float*)d_in[2];
  const float* wv = (const float*)d_in[3];
  const float* wo = (const float*)d_in[4];
  float* out = (float*)d_out;

  // Workspace carve-up (bf16). Total ~92.3 MB.
  char* w = (char*)d_ws;
  bf16* xb  = (bf16*)w; w += (size_t)Rows * Emb * 2;
  bf16* wqt = (bf16*)w; w += (size_t)Emb * Emb * 2;
  bf16* wkt = (bf16*)w; w += (size_t)Emb * Emb * 2;
  bf16* wvt = (bf16*)w; w += (size_t)Emb * Emb * 2;
  bf16* wot = (bf16*)w; w += (size_t)Emb * Emb * 2;
  bf16* Qm  = (bf16*)w; w += (size_t)Bsz * NH * Cctx * Dh * 2;
  bf16* Km  = (bf16*)w; w += (size_t)Bsz * NH * Cctx * Dh * 2;
  bf16* Vtm = (bf16*)w; w += (size_t)Bsz * NH * Dh * Cctx * 2;
  bf16* hid = (bf16*)w; w += (size_t)Rows * HD * 2;

  {
    const int nvec = Rows * Emb / 4;
    cvt_x_kernel<<<nvec / 256, 256, 0, stream>>>(x, xb, nvec);
  }
  {
    dim3 g(Emb / 16, Emb / 16, 4);
    cvt_w_kernel<<<g, 256, 0, stream>>>(wq, wk, wv, wo, wqt, wkt, wvt, wot);
  }
  {
    dim3 g(Rows / 256, HD / 64, 3);
    qkv_kernel<<<g, 256, 0, stream>>>(xb, wqt, wkt, wvt, Qm, Km, Vtm);
  }
  {
    dim3 g(Cctx / 128, NH, Bsz);
    attn_kernel<<<g, 256, 0, stream>>>(Qm, Km, Vtm, hid);
  }
  {
    dim3 g(Rows / 256, Emb / 64);
    oproj_kernel<<<g, 256, 0, stream>>>(hid, wot, out);
  }
}
